// MLALayer_88648124991545
// MI455X (gfx1250) — matrix operand order repaired and checked by reading, not compiled, not run on hardware
//
#include <hip/hip_runtime.h>

// ---------------------------------------------------------------------------
// MLA layer for MI455X (gfx1250): bf16 WMMA everywhere, fp32 accumulate.
// GEMMs use async global->LDS copies (ASYNCcnt) with double-buffered LDS.
// ---------------------------------------------------------------------------

#define USE_ASYNC_COPY 1

typedef __attribute__((ext_vector_type(16))) __bf16       v16bf;
typedef __attribute__((ext_vector_type(8)))  float        v8f;
typedef __attribute__((ext_vector_type(4)))  unsigned int u32x4;

union FragU { u32x4 u[2]; v16bf v; };   // 32 bytes = 8 VGPRs (16 bf16 / lane)

// Problem constants
constexpr int kB  = 2;
constexpr int kT  = 2048;
constexpr int kC  = 2048;
constexpr int kH  = 16;
constexpr int kHS = 128;
constexpr int kRD = 64;
constexpr int kR  = 512;           // KV_RANK == Q_RANK
constexpr int kD  = kHS + kRD;     // 192 (content || rope)
constexpr int kBT = kB * kT;       // 4096

__device__ __forceinline__ unsigned short f2bf(float f) {
  unsigned int u = __float_as_uint(f);
  u += 0x7fffu + ((u >> 16) & 1u);       // round-to-nearest-even
  return (unsigned short)(u >> 16);
}

__device__ __forceinline__ v8f wmma_bf16(v16bf a, v16bf b, v8f c) {
  // D = A(16x32 bf16) * B(32x16 bf16) + C(16x16 f32)
  return __builtin_amdgcn_wmma_f32_16x16x32_bf16(
      /*neg_a=*/false, a, /*neg_b=*/false, b,
      /*c_mod=*/(short)0, c, /*reuse_a=*/false, /*reuse_b=*/false);
}

// Low 32 bits of a generic pointer into LDS == workgroup-relative LDS offset.
__device__ __forceinline__ unsigned lds_off(const void* p) {
  return (unsigned)(unsigned long long)p;
}

// Per-lane async DMA: 16 bytes global -> LDS, tracked by ASYNCcnt.
__device__ __forceinline__ void async_copy_b128(unsigned lds, const void* g) {
#if USE_ASYNC_COPY
  asm volatile("global_load_async_to_lds_b128 %0, %1, off"
               :: "v"(lds), "v"((unsigned long long)g)
               : "memory");
#endif
}

__device__ __forceinline__ void wait_async_all() {
#if USE_ASYNC_COPY
  asm volatile("s_wait_asynccnt 0" ::: "memory");
#endif
}

// ---------------------------------------------------------------------------
// Elementwise helpers
// ---------------------------------------------------------------------------
__global__ void cvt_f32_to_bf16(const float* __restrict__ in,
                                unsigned short* __restrict__ out, int n) {
  int i = blockIdx.x * blockDim.x + threadIdx.x;
  if (i < n) out[i] = f2bf(in[i]);
}

// in: (K,N) fp32 row-major  ->  out: (N,K) bf16 row-major (i.e. B^T)
__global__ void transpose_cvt(const float* __restrict__ in,
                              unsigned short* __restrict__ out, int K, int N) {
  int i = blockIdx.x * blockDim.x + threadIdx.x;
  if (i >= K * N) return;
  int k = i / N;
  int n = i - k * N;
  out[(size_t)n * K + k] = f2bf(in[i]);
}

// up: (B*T, H*HS) fp32. Writes content half of K or Q buffer [(bh,t,0..127)],
// and (for the KV path) V transposed per head: Vt[(bh, d, t)].
__global__ void scatter_content(const float* __restrict__ up,
                                unsigned short* __restrict__ KQ,
                                unsigned short* __restrict__ Vt, int writeV) {
  int i = blockIdx.x * blockDim.x + threadIdx.x;
  if (i >= kBT * kH * kHS) return;
  int d = i & (kHS - 1);
  int h = (i >> 7) & (kH - 1);
  int t = (i >> 11) & (kT - 1);
  int b = i >> 22;
  unsigned short w = f2bf(up[i]);
  size_t bh = (size_t)(b * kH + h);
  KQ[(bh * kT + t) * kD + d] = w;
  if (writeV) Vt[(bh * kHS + d) * kT + t] = w;
}

// xr: (B*T, H*RD) fp32 pre-rope. Applies rope (cos/sin are (T,RD)) and writes
// into the rope half of K or Q buffer: dst[(bh, t, 128 + d)].
__global__ void rope_scatter(const float* __restrict__ xr,
                             const float* __restrict__ fcos,
                             const float* __restrict__ fsin,
                             unsigned short* __restrict__ dst) {
  int i = blockIdx.x * blockDim.x + threadIdx.x;
  if (i >= kBT * kH * kRD) return;
  int d = i & (kRD - 1);
  int h = (i >> 6) & (kH - 1);
  int t = (i >> 10) & (kT - 1);
  int b = i >> 21;
  float xv = xr[i];
  float other = (d < kRD / 2) ? -xr[i + kRD / 2] : xr[i - kRD / 2];
  float c = fcos[t * kRD + d];
  float s = fsin[t * kRD + d];
  size_t bh = (size_t)(b * kH + h);
  dst[(bh * kT + t) * kD + kHS + d] = f2bf(xv * c + other * s);
}

// ---------------------------------------------------------------------------
// bf16 WMMA GEMM:  C(MxN) = A(MxK, bf16 row-major) * Bt(NxK, bf16 row-major)^T
// Block tile 128x64, 8 waves, each wave 2x2 fragments of 16x16, K-step 32.
// Double-buffered LDS fed by async global->LDS DMA (ASYNCcnt).
// Sync protocol per iteration:
//   s_wait_asynccnt 0   (own DMAs done)
//   __syncthreads()     (=> everyone's DMAs done; prev-buffer reads also done)
//   issue DMA for tile k0+32 into the other buffer
//   compute from current buffer
// ---------------------------------------------------------------------------
template <bool OUT_BF16>
__global__ __launch_bounds__(256) void gemm_bf16(
    const unsigned short* __restrict__ A,
    const unsigned short* __restrict__ Bt,
    void* __restrict__ Cout, int M, int N, int K) {
  constexpr int LDA = 40;  // padded halfword stride (80B, multiple of 16B)
  __shared__ unsigned short As[2][128 * LDA];
  __shared__ unsigned short Bs[2][64 * LDA];

  const int tid  = threadIdx.x;
  const int lane = tid & 31;
  const int wave = tid >> 5;
  const int m0 = blockIdx.y * 128;
  const int n0 = blockIdx.x * 64;
  const int wm = (wave & 3) * 32;
  const int wn = (wave >> 2) * 32;

  // Per-thread staging coordinates (b128 granules)
  const int rB  = tid >> 2;
  const int kB8 = (tid & 3) * 8;

  auto issue_tile = [&](int k0, int buf) {
    // A tile: 128x32 bf16 = 512 granules, 2 per thread
#pragma unroll
    for (int i = 0; i < 2; ++i) {
      int e = tid + i * 256;
      int row = e >> 2;
      int k8 = (e & 3) * 8;
      async_copy_b128(lds_off(&As[buf][row * LDA + k8]),
                      A + (size_t)(m0 + row) * K + k0 + k8);
#if !USE_ASYNC_COPY
      *reinterpret_cast<u32x4*>(&As[buf][row * LDA + k8]) =
          *reinterpret_cast<const u32x4*>(A + (size_t)(m0 + row) * K + k0 + k8);
#endif
    }
    // B^T tile: 64x32 bf16 = 256 granules, 1 per thread
    async_copy_b128(lds_off(&Bs[buf][rB * LDA + kB8]),
                    Bt + (size_t)(n0 + rB) * K + k0 + kB8);
#if !USE_ASYNC_COPY
    *reinterpret_cast<u32x4*>(&Bs[buf][rB * LDA + kB8]) =
        *reinterpret_cast<const u32x4*>(Bt + (size_t)(n0 + rB) * K + k0 + kB8);
#endif
  };

  v8f zero = {0.f, 0.f, 0.f, 0.f, 0.f, 0.f, 0.f, 0.f};
  v8f acc[2][2];
#pragma unroll
  for (int i = 0; i < 2; ++i)
#pragma unroll
    for (int j = 0; j < 2; ++j) acc[i][j] = zero;

  const int kbA = (lane < 16) ? 0 : 8;
  const int kbB = (lane < 16) ? 0 : 8;

  issue_tile(0, 0);
  int buf = 0;
  for (int k0 = 0; k0 < K; k0 += 32, buf ^= 1) {
    wait_async_all();
    __syncthreads();
    if (k0 + 32 < K) issue_tile(k0 + 32, buf ^ 1);

    // A fragments (ISA 16-bit 16x32 layout): lane = M row,
    // lanes 0-15 hold K {kb..kb+7, kb+16..kb+23} with kb=0; lanes 16-31 kb=8.
    FragU aF[2], bF[2];
#pragma unroll
    for (int f = 0; f < 2; ++f) {
      int arow = wm + f * 16 + (lane & 15);
      aF[f].u[0] = *reinterpret_cast<const u32x4*>(&As[buf][arow * LDA + kbA]);
      aF[f].u[1] = *reinterpret_cast<const u32x4*>(&As[buf][arow * LDA + kbA + 16]);
      int brow = wn + f * 16 + (lane & 15);
      bF[f].u[0] = *reinterpret_cast<const u32x4*>(&Bs[buf][brow * LDA + kbB]);
      bF[f].u[1] = *reinterpret_cast<const u32x4*>(&Bs[buf][brow * LDA + kbB + 16]);
    }
#pragma unroll
    for (int fm = 0; fm < 2; ++fm)
#pragma unroll
      for (int fn = 0; fn < 2; ++fn)
        acc[fm][fn] = wmma_bf16(aF[fm].v, bF[fn].v, acc[fm][fn]);
#if !USE_ASYNC_COPY
    __syncthreads();  // sync-path needs a trailing barrier before restaging
#endif
  }

  // Store C (layout: VGPR r -> rows r / r+8; lane%16 -> column)
  const int colb = lane & 15;
  const int rb = (lane < 16) ? 0 : 8;
#pragma unroll
  for (int fm = 0; fm < 2; ++fm)
#pragma unroll
    for (int fn = 0; fn < 2; ++fn)
#pragma unroll
      for (int r = 0; r < 8; ++r) {
        size_t row = (size_t)(m0 + wm + fm * 16 + rb + r);
        int col = n0 + wn + fn * 16 + colb;
        float v = acc[fm][fn][r];
        if (OUT_BF16)
          ((unsigned short*)Cout)[row * N + col] = f2bf(v);
        else
          ((float*)Cout)[row * N + col] = v;
      }
}

// ---------------------------------------------------------------------------
// Flash attention: one wave per 16-query tile, 32 keys per step.
// Q/K: (bh, T, 192) bf16.  Vt: (bh, 128, T) bf16.  Y: (B*T, 2048) bf16.
// ---------------------------------------------------------------------------
__global__ __launch_bounds__(256) void mla_attention(
    const unsigned short* __restrict__ Qbuf,
    const unsigned short* __restrict__ Kbuf,
    const unsigned short* __restrict__ Vt,
    unsigned short* __restrict__ Y) {
  __shared__ unsigned short Pbuf[8][16 * 32];  // per-wave P staging (1KB each)

  const int tid  = threadIdx.x;
  const int lane = tid & 31;
  const int wave = tid >> 5;
  const int bh = blockIdx.x;           // b*H + h
  const int b  = bh >> 4;
  const int h  = bh & 15;
  const int qt = blockIdx.y * 8 + wave;
  const int q0 = qt * 16;

  const size_t qkbase = (size_t)bh * kT * kD;
  const size_t vbase  = (size_t)bh * kHS * kT;

  const int kbA = (lane < 16) ? 0 : 8;    // A-fragment K base
  const int kbB = (lane < 16) ? 0 : 8;    // B-fragment K base
  const int col = lane & 15;
  const int rb  = (lane < 16) ? 0 : 8;

  // Load Q fragments once: 6 chunks of 16x32 over d=192
  FragU qF[6];
  {
    int arow = q0 + col;
#pragma unroll
    for (int c = 0; c < 6; ++c) {
      const unsigned short* p = Qbuf + qkbase + (size_t)arow * kD + c * 32 + kbA;
      qF[c].u[0] = *reinterpret_cast<const u32x4*>(p);
      qF[c].u[1] = *reinterpret_cast<const u32x4*>(p + 16);
    }
  }

  v8f zero = {0.f, 0.f, 0.f, 0.f, 0.f, 0.f, 0.f, 0.f};
  v8f o[8];
  float mrow[8], lrow[8];
#pragma unroll
  for (int f = 0; f < 8; ++f) o[f] = zero;
#pragma unroll
  for (int r = 0; r < 8; ++r) { mrow[r] = -1e30f; lrow[r] = 0.f; }

  const float scale = 0.08838834764831845f;  // HS^-0.5

  for (int kt = 0; kt < q0 + 16; kt += 32) {
    // ---- S = Q K^T for two 16-key subtiles ----
    v8f s0 = zero, s1 = zero;
    const int krow = kt + col;
#pragma unroll
    for (int c = 0; c < 6; ++c) {
      FragU kf;
      const unsigned short* p = Kbuf + qkbase + (size_t)krow * kD + c * 32 + kbB;
      kf.u[0] = *reinterpret_cast<const u32x4*>(p);
      kf.u[1] = *reinterpret_cast<const u32x4*>(p + 16);
      s0 = wmma_bf16(qF[c].v, kf.v, s0);
      const unsigned short* p2 = p + (size_t)16 * kD;
      kf.u[0] = *reinterpret_cast<const u32x4*>(p2);
      kf.u[1] = *reinterpret_cast<const u32x4*>(p2 + 16);
      s1 = wmma_bf16(qF[c].v, kf.v, s1);
    }

    // ---- online softmax (per C-layout row r -> global row q0+rb+r) ----
#pragma unroll
    for (int r = 0; r < 8; ++r) {
      int q = q0 + rb + r;
      float a0 = s0[r] * scale;
      float a1 = s1[r] * scale;
      if (kt + col > q)       a0 = -1e30f;   // causal mask
      if (kt + 16 + col > q)  a1 = -1e30f;
      float rmax = fmaxf(a0, a1);
      rmax = fmaxf(rmax, __shfl_xor(rmax, 1, 32));
      rmax = fmaxf(rmax, __shfl_xor(rmax, 2, 32));
      rmax = fmaxf(rmax, __shfl_xor(rmax, 4, 32));
      rmax = fmaxf(rmax, __shfl_xor(rmax, 8, 32));
      float mnew  = fmaxf(mrow[r], rmax);
      float alpha = __expf(mrow[r] - mnew);
      float p0 = __expf(a0 - mnew);
      float p1 = __expf(a1 - mnew);
      float rsum = p0 + p1;
      rsum += __shfl_xor(rsum, 1, 32);
      rsum += __shfl_xor(rsum, 2, 32);
      rsum += __shfl_xor(rsum, 4, 32);
      rsum += __shfl_xor(rsum, 8, 32);
      lrow[r] = lrow[r] * alpha + rsum;
      mrow[r] = mnew;
#pragma unroll
      for (int f = 0; f < 8; ++f) o[f][r] *= alpha;
      // C-layout -> LDS (row-major 16x32), for A-layout reload
      int m = rb + r;
      Pbuf[wave][m * 32 + col]      = f2bf(p0);
      Pbuf[wave][m * 32 + 16 + col] = f2bf(p1);
    }

    // ---- reload P as A fragment (same-wave LDS ops are in-order) ----
    FragU pf;
    {
      const unsigned short* pp = &Pbuf[wave][(lane & 15) * 32 + kbA];
      pf.u[0] = *reinterpret_cast<const u32x4*>(pp);
      pf.u[1] = *reinterpret_cast<const u32x4*>(pp + 16);
    }

    // ---- O += P @ V  (B fragments contiguous from Vt) ----
#pragma unroll
    for (int f = 0; f < 8; ++f) {
      FragU vf;
      const unsigned short* vp =
          Vt + vbase + (size_t)(f * 16 + col) * kT + kt + kbB;
      vf.u[0] = *reinterpret_cast<const u32x4*>(vp);
      vf.u[1] = *reinterpret_cast<const u32x4*>(vp + 16);
      o[f] = wmma_bf16(pf.v, vf.v, o[f]);
    }
  }

  // ---- normalize, write y[(b*T + t), h*128 + d] as bf16 ----
#pragma unroll
  for (int r = 0; r < 8; ++r) {
    float inv = 1.0f / lrow[r];
    int t = q0 + rb + r;
    size_t yrow = ((size_t)b * kT + t) * (size_t)(kH * kHS) + h * kHS;
#pragma unroll
    for (int f = 0; f < 8; ++f)
      Y[yrow + f * 16 + col] = f2bf(o[f][r] * inv);
  }
}

// ---------------------------------------------------------------------------
// Host launcher
// ---------------------------------------------------------------------------
extern "C" void kernel_launch(void* const* d_in, const int* in_sizes, int n_in,
                              void* d_out, int out_size, void* d_ws,
                              size_t ws_size, hipStream_t stream) {
  const float* x    = (const float*)d_in[0];
  const float* fc   = (const float*)d_in[1];
  const float* fs   = (const float*)d_in[2];
  const float* wkvd = (const float*)d_in[3];
  const float* wkvu = (const float*)d_in[4];
  const float* wkr  = (const float*)d_in[5];
  const float* wqd  = (const float*)d_in[6];
  const float* wqu  = (const float*)d_in[7];
  const float* wqr  = (const float*)d_in[8];
  const float* wout = (const float*)d_in[9];
  float* out = (float*)d_out;

  char* ws = (char*)d_ws;
  size_t off = 0;
  auto alloc = [&](size_t bytes) -> void* {
    void* p = ws + off;
    off += (bytes + 255) & ~(size_t)255;
    return p;
  };

  unsigned short* xbf   = (unsigned short*)alloc((size_t)kBT * kC * 2);
  unsigned short* wkvdT = (unsigned short*)alloc((size_t)kC * kR * 2);
  unsigned short* wkvuT = (unsigned short*)alloc((size_t)kR * kC * 2);
  unsigned short* wkrT  = (unsigned short*)alloc((size_t)kR * kH * kRD * 2);
  unsigned short* wqdT  = (unsigned short*)alloc((size_t)kC * kR * 2);
  unsigned short* wquT  = (unsigned short*)alloc((size_t)kR * kC * 2);
  unsigned short* wqrT  = (unsigned short*)alloc((size_t)kR * kH * kRD * 2);
  unsigned short* woutT = (unsigned short*)alloc((size_t)kC * kC * 2);
  unsigned short* latkv = (unsigned short*)alloc((size_t)kBT * kR * 2);
  unsigned short* latq  = (unsigned short*)alloc((size_t)kBT * kR * 2);
  float*          upf   = (float*)alloc((size_t)kBT * kC * 4);        // reused
  float*          ropef = (float*)alloc((size_t)kBT * kH * kRD * 4);  // reused
  unsigned short* Qbuf  = (unsigned short*)alloc((size_t)kB * kH * kT * kD * 2);
  unsigned short* Kbuf  = (unsigned short*)alloc((size_t)kB * kH * kT * kD * 2);
  unsigned short* Vt    = (unsigned short*)alloc((size_t)kB * kH * kHS * kT * 2);
  unsigned short* ybf   = (unsigned short*)alloc((size_t)kBT * kC * 2);

  const int TPB = 256;
  auto nb = [](long n) { return (unsigned)((n + 255) / 256); };

  // bf16 conversions / weight transposes
  cvt_f32_to_bf16<<<nb((long)kBT * kC), TPB, 0, stream>>>(x, xbf, kBT * kC);
  transpose_cvt<<<nb((long)kC * kR), TPB, 0, stream>>>(wkvd, wkvdT, kC, kR);
  transpose_cvt<<<nb((long)kR * kC), TPB, 0, stream>>>(wkvu, wkvuT, kR, kC);
  transpose_cvt<<<nb((long)kR * kH * kRD), TPB, 0, stream>>>(wkr, wkrT, kR, kH * kRD);
  transpose_cvt<<<nb((long)kC * kR), TPB, 0, stream>>>(wqd, wqdT, kC, kR);
  transpose_cvt<<<nb((long)kR * kC), TPB, 0, stream>>>(wqu, wquT, kR, kC);
  transpose_cvt<<<nb((long)kR * kH * kRD), TPB, 0, stream>>>(wqr, wqrT, kR, kH * kRD);
  transpose_cvt<<<nb((long)kC * kC), TPB, 0, stream>>>(wout, woutT, kC, kC);

  dim3 gDown(kR / 64, kBT / 128);            // N=512
  dim3 gUp(kC / 64, kBT / 128);              // N=2048
  dim3 gRope((kH * kRD) / 64, kBT / 128);    // N=1024

  // KV path
  gemm_bf16<true ><<<gDown, 256, 0, stream>>>(xbf,   wkvdT, latkv, kBT, kR, kC);
  gemm_bf16<false><<<gUp,   256, 0, stream>>>(latkv, wkvuT, upf,   kBT, kC, kR);
  scatter_content<<<nb((long)kBT * kC), TPB, 0, stream>>>(upf, Kbuf, Vt, 1);
  gemm_bf16<false><<<gRope, 256, 0, stream>>>(latkv, wkrT,  ropef, kBT, kH * kRD, kR);
  rope_scatter<<<nb((long)kBT * kH * kRD), TPB, 0, stream>>>(ropef, fc, fs, Kbuf);

  // Q path
  gemm_bf16<true ><<<gDown, 256, 0, stream>>>(xbf,  wqdT, latq,  kBT, kR, kC);
  gemm_bf16<false><<<gUp,   256, 0, stream>>>(latq, wquT, upf,   kBT, kC, kR);
  scatter_content<<<nb((long)kBT * kC), TPB, 0, stream>>>(upf, Qbuf, Vt, 0);
  gemm_bf16<false><<<gRope, 256, 0, stream>>>(latq, wqrT, ropef, kBT, kH * kRD, kR);
  rope_scatter<<<nb((long)kBT * kH * kRD), TPB, 0, stream>>>(ropef, fc, fs, Qbuf);

  // Flash attention
  dim3 ga(kB * kH, kT / 128);
  mla_attention<<<ga, 256, 0, stream>>>(Qbuf, Kbuf, Vt, ybf);

  // Output projection (fp32 to d_out)
  gemm_bf16<false><<<gUp, 256, 0, stream>>>(ybf, woutT, out, kBT, kC, kC);
}
